// SequenceHMM_26757646254536
// MI455X (gfx1250) — matrix operand order repaired and checked by reading, not compiled, not run on hardware
//
#include <hip/hip_runtime.h>
#include <hip/hip_bf16.h>

typedef __attribute__((ext_vector_type(16))) _Float16 v16h;
typedef __attribute__((ext_vector_type(8)))  float    v8f;

#define S  512
#define VV 256

// ---------------------------------------------------------------------------
// union to build WMMA 16-bit operands from raw 128-bit loads
// ---------------------------------------------------------------------------
union AB16 {
    v16h  v;
    uint4 q[2];
};

// order-preserving float <-> uint key (IEEE total order) for LDS u32 atomicMax
__device__ __forceinline__ unsigned fkey(float f) {
    unsigned u = __float_as_uint(f);
    return (u & 0x80000000u) ? ~u : (u | 0x80000000u);
}
__device__ __forceinline__ float funkey(unsigned k) {
    return (k & 0x80000000u) ? __uint_as_float(k & 0x7fffffffu)
                             : __uint_as_float(~k);
}

// ---------------------------------------------------------------------------
// Kernel 1: row-softmax of state_transitions -> P, written directly in the
// WMMA B-operand swizzled layout.
//   kt=k>>5, kin=k&31, nt=n>>4, nin=n&15
//   lane = nin + 16*((kin>>3)&1); half = (kin&7) | ((kin&16)>>1)
//   idx  = (((kt*32)+nt)*32 + lane)*16 + half
// ---------------------------------------------------------------------------
__global__ void hmm_prep_trans(const float* __restrict__ trans,
                               _Float16* __restrict__ ppack) {
    __shared__ float red[256];
    const int i  = blockIdx.x;        // from-state = K index
    const int t0 = threadIdx.x;

    float a = trans[i * S + t0];
    float b = trans[i * S + t0 + 256];

    red[t0] = fmaxf(a, b);
    __syncthreads();
    for (int off = 128; off > 0; off >>= 1) {
        if (t0 < off) red[t0] = fmaxf(red[t0], red[t0 + off]);
        __syncthreads();
    }
    float mx = red[0];
    __syncthreads();

    float ea = expf(a - mx), eb = expf(b - mx);
    red[t0] = ea + eb;
    __syncthreads();
    for (int off = 128; off > 0; off >>= 1) {
        if (t0 < off) red[t0] += red[t0 + off];
        __syncthreads();
    }
    float inv = 1.0f / red[0];

    const int kt   = i >> 5, kin = i & 31;
    const int half = (kin & 7) | ((kin & 16) >> 1);
    const int lhi  = (kin & 8) ? 16 : 0;
    {   // j = t0
        int nt = t0 >> 4, nin = t0 & 15;
        long idx = (((long)(kt * 32 + nt)) * 32 + (nin + lhi)) * 16 + half;
        ppack[idx] = (_Float16)(ea * inv);
    }
    {   // j = t0 + 256
        int j = t0 + 256;
        int nt = j >> 4, nin = j & 15;
        long idx = (((long)(kt * 32 + nt)) * 32 + (nin + lhi)) * 16 + half;
        ppack[idx] = (_Float16)(eb * inv);
    }
}

// ---------------------------------------------------------------------------
// Kernel 2: row log-softmax of observation_transitions [S][V], stored
// transposed as obs_logT[V][S] so the scan's per-state gather is coalesced.
// ---------------------------------------------------------------------------
__global__ void hmm_prep_obs(const float* __restrict__ obsmat,
                             float* __restrict__ obs_logT) {
    __shared__ float red[256];
    const int j = blockIdx.x;         // state row
    const int o = threadIdx.x;        // vocab col

    float a = obsmat[j * VV + o];
    red[o] = a;
    __syncthreads();
    for (int off = 128; off > 0; off >>= 1) {
        if (o < off) red[o] = fmaxf(red[o], red[o + off]);
        __syncthreads();
    }
    float mx = red[0];
    __syncthreads();
    red[o] = expf(a - mx);
    __syncthreads();
    for (int off = 128; off > 0; off >>= 1) {
        if (o < off) red[o] += red[o + off];
        __syncthreads();
    }
    float lse = mx + logf(red[0]);
    obs_logT[o * S + j] = a - lse;
}

// ---------------------------------------------------------------------------
// Kernel 3: log-softmax of prior_states[512]
// ---------------------------------------------------------------------------
__global__ void hmm_prep_prior(const float* __restrict__ prior,
                               float* __restrict__ prior_log) {
    __shared__ float red[512];
    const int t = threadIdx.x;
    float a = prior[t];
    red[t] = a;
    __syncthreads();
    for (int off = 256; off > 0; off >>= 1) {
        if (t < off) red[t] = fmaxf(red[t], red[t + off]);
        __syncthreads();
    }
    float mx = red[0];
    __syncthreads();
    red[t] = expf(a - mx);
    __syncthreads();
    for (int off = 256; off > 0; off >>= 1) {
        if (t < off) red[t] += red[t + off];
        __syncthreads();
    }
    float lse = mx + logf(red[0]);
    prior_log[t] = a - lse;
}

// ---------------------------------------------------------------------------
// Kernel 4: persistent forward scan. 2 blocks x 1024 threads (32 waves).
// Wave w owns the 16x16 tile for states [16w,16w+16).
// Invariant: X[m][i] = exp(alpha[m][i] - carry[m])  (f16 in LDS).
// 2 barriers per timestep; y never touches LDS.
// ---------------------------------------------------------------------------
#define XSTR 520   // f16 row stride: pad kills bank conflicts, keeps 16B align

__global__ void __launch_bounds__(1024)
hmm_scan(const _Float16* __restrict__ ppack,
         const float* __restrict__ obs_logT,
         const float* __restrict__ prior_log,
         const int* __restrict__ observations,   // [B][T]
         float* __restrict__ out, int T) {
    __shared__ _Float16 X[16 * XSTR] __attribute__((aligned(16)));
    __shared__ unsigned rowkey[2][16];   // double-buffered row-max keys
    __shared__ float    carry[16];
    __shared__ int      sobs[16];

    const int tid  = threadIdx.x;
    const int wave = tid >> 5;
    const int lane = tid & 31;
    const int b0   = blockIdx.x * 16;
    const int m16  = lane & 15;
    const int hb   = (lane & 16) ? 8 : 0;   // A-operand K sub-block per half
    const int mb   = hb;                    // C/D row base per half
    const uint4* pq = (const uint4*)ppack;

    if (tid < 16)  sobs[tid] = observations[(b0 + tid) * T];
    if (tid < 32)  rowkey[tid >> 4][tid & 15] = 0u;
    __syncthreads();

    // ---- t = 0 : y0 = emis0 + prior (generic 8-per-thread mapping) ----
    {
        float y0[8];
        #pragma unroll
        for (int e = 0; e < 8; ++e) {
            int idx = tid + e * 1024;
            int m = idx >> 9, j = idx & 511;
            float y = obs_logT[sobs[m] * S + j] + prior_log[j];
            y0[e] = y;
            atomicMax(&rowkey[0][m], fkey(y));
        }
        __syncthreads();
        #pragma unroll
        for (int e = 0; e < 8; ++e) {
            int idx = tid + e * 1024;
            int m = idx >> 9, j = idx & 511;
            X[m * XSTR + j] = (_Float16)expf(y0[e] - funkey(rowkey[0][m]));
        }
        if (tid < 16) carry[tid] = funkey(rowkey[0][tid]);
        if (tid >= 16 && tid < 32) rowkey[1][tid - 16] = 0u;
        if (tid >= 32 && tid < 48) {
            int m = tid - 32;
            sobs[m] = observations[(b0 + m) * T + (T > 1 ? 1 : 0)];
        }
        __syncthreads();
    }

    const int j = wave * 16 + m16;   // this lane's output column

    for (int t = 1; t < T; ++t) {
        const unsigned cur = (unsigned)(t & 1), nxt = cur ^ 1u;

        // ================= phase A: matmul + y + row-max =================
        // emission gathers issued first: L2 latency hides under the WMMAs
        float em[8];
        #pragma unroll
        for (int v = 0; v < 8; ++v)
            em[v] = obs_logT[sobs[v + mb] * S + j];

        // double-buffered A operand, dual accumulator chains
        AB16 ab[2];
        {
            const int ha = m16 * XSTR + hb;
            ab[0].q[0] = *(const uint4*)&X[ha];
            ab[0].q[1] = *(const uint4*)&X[ha + 16];
        }
        v8f c0 = {0.f, 0.f, 0.f, 0.f, 0.f, 0.f, 0.f, 0.f};
        v8f c1 = {0.f, 0.f, 0.f, 0.f, 0.f, 0.f, 0.f, 0.f};
        #pragma unroll
        for (int kt = 0; kt < 16; ++kt) {
            if (kt < 15) {                         // prefetch next A tile
                const int ha = m16 * XSTR + (kt + 1) * 32 + hb;
                AB16& an = ab[(kt + 1) & 1];
                an.q[0] = *(const uint4*)&X[ha];
                an.q[1] = *(const uint4*)&X[ha + 16];
            }
            AB16 b;
            const long bi = (((long)(kt * 32 + wave)) * 32 + lane) * 2;
            b.q[0] = pq[bi];
            b.q[1] = pq[bi + 1];
            if (kt & 1)
                c1 = __builtin_amdgcn_wmma_f32_16x16x32_f16(
                        false, ab[1].v, false, b.v, (short)0, c1, false, false);
            else
                c0 = __builtin_amdgcn_wmma_f32_16x16x32_f16(
                        false, ab[0].v, false, b.v, (short)0, c0, false, false);
        }

        float y[8];
        #pragma unroll
        for (int v = 0; v < 8; ++v)
            y[v] = em[v] + logf(c0[v] + c1[v]);

        // per-row max: butterfly over the 16-lane half, one atomic per row
        #pragma unroll
        for (int v = 0; v < 8; ++v) {
            float m = y[v];
            m = fmaxf(m, __shfl_xor(m, 1));
            m = fmaxf(m, __shfl_xor(m, 2));
            m = fmaxf(m, __shfl_xor(m, 4));
            m = fmaxf(m, __shfl_xor(m, 8));
            if (m16 == v) atomicMax(&rowkey[cur][v + mb], fkey(m));
        }
        __syncthreads();

        // ================= phase B: renormalize + housekeeping ============
        #pragma unroll
        for (int v = 0; v < 8; ++v) {
            float rmax = funkey(rowkey[cur][v + mb]);
            X[(v + mb) * XSTR + j] = (_Float16)expf(y[v] - rmax);
        }
        if (tid < 16) carry[tid] += funkey(rowkey[cur][tid]);
        if (tid >= 16 && tid < 32) rowkey[nxt][tid - 16] = 0u;
        if (tid >= 32 && tid < 48) {              // prefetch next observation
            int m  = tid - 32;
            int tn = (t + 1 < T) ? t + 1 : t;
            sobs[m] = observations[(b0 + m) * T + tn];
        }
        __syncthreads();
    }

    // ---- output: out[b] = carry + log(sum_j X[b][j]) ----
    if (tid < 512) {
        int m = tid >> 5, j0 = tid & 31;
        float sm = 0.0f;
        for (int jj = j0; jj < 512; jj += 32) sm += (float)X[m * XSTR + jj];
        for (int o = 16; o > 0; o >>= 1) sm += __shfl_xor(sm, o);
        if (j0 == 0) out[b0 + m] = carry[m] + logf(sm);
    }
}

// ---------------------------------------------------------------------------
// Host launcher
// ---------------------------------------------------------------------------
extern "C" void kernel_launch(void* const* d_in, const int* in_sizes, int n_in,
                              void* d_out, int out_size, void* d_ws, size_t ws_size,
                              hipStream_t stream) {
    const float* trans        = (const float*)d_in[0];   // [512,512]
    const float* obsmat       = (const float*)d_in[1];   // [512,256]
    const float* prior        = (const float*)d_in[2];   // [512]
    const int*   observations = (const int*)d_in[3];     // [32,T]
    float*       out          = (float*)d_out;           // [32]

    const int B = 32;
    const int T = in_sizes[3] / B;

    char*     ws        = (char*)d_ws;
    _Float16* ppack     = (_Float16*)ws;                  // 512*512*2 = 512 KB
    float*    obs_logT  = (float*)(ws + 512 * 1024);      // 256*512*4 = 512 KB
    float*    prior_log = (float*)(ws + 1024 * 1024);     // 2 KB

    hmm_prep_trans<<<512, 256, 0, stream>>>(trans, ppack);
    hmm_prep_obs  <<<512, 256, 0, stream>>>(obsmat, obs_logT);
    hmm_prep_prior<<<1,   512, 0, stream>>>(prior, prior_log);
    hmm_scan      <<<2,  1024, 0, stream>>>(ppack, obs_logT, prior_log,
                                            observations, out, T);
}
